// ODEFunc_67053029425655
// MI455X (gfx1250) — compile-verified, matrix operand order checked by reading, not run on hardware
//
#include <hip/hip_runtime.h>

// CDNA5 / gfx1250: wave32, WMMA 16x16x32 f16 -> f32
typedef __attribute__((ext_vector_type(16))) _Float16 v16h;
typedef __attribute__((ext_vector_type(8)))  float    v8f;
typedef __attribute__((ext_vector_type(2)))  _Float16 h2f;   // packed f16 pair

#define NP 5

static __device__ __forceinline__ int bperm_i(int srcLane, int v) {
  return __builtin_amdgcn_ds_bpermute(srcLane << 2, v);
}
static __device__ __forceinline__ float bperm_f(int srcLane, float v) {
  union { float f; int i; } a, b;
  a.f = v;
  b.i = __builtin_amdgcn_ds_bpermute(srcLane << 2, a.i);
  return b.f;
}
static __device__ __forceinline__ h2f relu2(h2f t) {
  const h2f z = {(_Float16)0.0f, (_Float16)0.0f};
  return __builtin_elementwise_max(t, z);   // v_pk_max_num_f16
}

union H2U { h2f h; unsigned u; };

__global__ __launch_bounds__(256) void odefunc_kernel(
    const float* __restrict__ y,
    const float* __restrict__ W1,  const float* __restrict__ b1,
    const float* __restrict__ w2,  const float* __restrict__ b2,
    const float* __restrict__ W3a, const float* __restrict__ b3a,
    const float* __restrict__ W3b, const float* __restrict__ b3b,
    const float* __restrict__ W3c, const float* __restrict__ b3c,
    float* __restrict__ out, int nB)
{
  const int  tid    = blockIdx.x * blockDim.x + threadIdx.x;
  const int  lane   = threadIdx.x & 31;
  const int  m      = lane & 15;       // fragment row/column index
  const bool hiHalf = lane >= 16;
  const bool inb    = tid < nB;

  // ---- per-element state: 5 positions + 5 velocities (2D) ----
  float Xx[NP], Xy[NP], Vx[NP], Vy[NP];
  if (inb) {
    const float4* yv = (const float4*)(y + (size_t)tid * 20);  // 80B row, 16B aligned
    float4 r0 = yv[0], r1 = yv[1], r2 = yv[2], r3 = yv[3], r4 = yv[4];
    Xx[0]=r0.x; Xy[0]=r0.y; Xx[1]=r0.z; Xy[1]=r0.w;
    Xx[2]=r1.x; Xy[2]=r1.y; Xx[3]=r1.z; Xy[3]=r1.w;
    Xx[4]=r2.x; Xy[4]=r2.y; Vx[0]=r2.z; Vy[0]=r2.w;
    Vx[1]=r3.x; Vy[1]=r3.y; Vx[2]=r3.z; Vy[2]=r3.w;
    Vx[3]=r4.x; Vy[3]=r4.y; Vx[4]=r4.z; Vy[4]=r4.w;
  } else {
    #pragma unroll
    for (int p = 0; p < NP; ++p) {     // distinct filler -> d2>0, no NaN into WMMA
      Xx[p] = (float)(p + 1); Xy[p] = 0.f; Vx[p] = 0.f; Vy[p] = 0.f;
    }
  }

  // ---- tiny weights (broadcast reads, L2-resident) ----
  float W1r[8], b1r[2], b3br[8], W3cr[8];
  #pragma unroll
  for (int j = 0; j < 8; ++j)  W1r[j]  = W1[j];
  b1r[0] = b1[0]; b1r[1] = b1[1];
  const float w2r = w2[0], b2r = b2[0], b3cr = b3c[0];
  #pragma unroll
  for (int j = 0; j < 8; ++j)  { b3br[j] = b3b[j]; W3cr[j] = W3c[j]; }

  // ---- lane-selected packed layer-1 weights: K-slice kb..kb+7 of W3a/b3a ----
  // (low lanes supply K=0..7 / K=16..23; high lanes K=8..15 / K=24..31 —
  //  both K-regions map to the SAME hidden-index slice kb..kb+7)
  const int kb = hiHalf ? 8 : 0;
  H2U WaS[4], baS[4];
  #pragma unroll
  for (int j = 0; j < 4; ++j) {
    WaS[j].h[0] = (_Float16)W3a[kb + 2*j];  WaS[j].h[1] = (_Float16)W3a[kb + 2*j + 1];
    baS[j].h[0] = (_Float16)b3a[kb + 2*j];  baS[j].h[1] = (_Float16)b3a[kb + 2*j + 1];
  }

  // ---- Combined A-fragment: two copies of W3b, fully utilizing 16x32 ----
  // rows 0..7  x K[0..15]  = W3b  (pair p0 of a block)
  // rows 8..15 x K[16..31] = W3b  (pair p1 of a block)
  v16h afragC;
  #pragma unroll
  for (int j = 0; j < 16; ++j) afragC[j] = (_Float16)0.0f;
  {
    const int row = m & 7;
    #pragma unroll
    for (int j = 0; j < 8; ++j) {
      const _Float16 w = (_Float16)W3b[row * 16 + kb + j];
      // halves 0..7 <-> K 0..15 region; halves 8..15 <-> K 16..31 region
      afragC[j]     = (m < 8) ? w : (_Float16)0.0f;
      afragC[8 + j] = (m < 8) ? (_Float16)0.0f : w;
    }
  }

  // ---- C-fragment: bias b3b on ALL rows (rows 0-7 = p0, rows 8-15 = p1) ----
  v8f cb;
  #pragma unroll
  for (int v = 0; v < 8; ++v) cb[v] = b3br[v];

  // ---- f3: 10 symmetric pairs as 5 pair-blocks, 2 pairs per WMMA ----
  static const int PI0[5] = {0, 0, 1, 1, 2}, PK0[5] = {1, 3, 2, 4, 4};
  static const int PI1[5] = {0, 0, 1, 2, 3}, PK1[5] = {2, 4, 3, 3, 4};

  float f3x[NP] = {0,0,0,0,0}, f3y[NP] = {0,0,0,0,0};
  #pragma unroll
  for (int blk = 0; blk < 5; ++blk) {
    const int i0 = PI0[blk], k0 = PK0[blk];
    const int i1 = PI1[blk], k1 = PK1[blk];

    // distances for both pairs of this block (own element)
    const float dx0 = Xx[i0] - Xx[k0], dy0 = Xy[i0] - Xy[k0];
    const float dx1 = Xx[i1] - Xx[k1], dy1 = Xy[i1] - Xy[k1];
    const float q0 = fmaf(dx0, dx0, dy0 * dy0);
    const float q1 = fmaf(dx1, dx1, dy1 * dy1);
    const float rinv0 = __frsqrt_rn(q0), rinv1 = __frsqrt_rn(q1);
    const float dA = q0 * rinv0, dB = q1 * rinv1;

    // pack both distances into one u32; fetch column-owners' packed d
    H2U dpk; dpk.h[0] = (_Float16)dA; dpk.h[1] = (_Float16)dB;
    H2U e0, e1;
    e0.u = (unsigned)bperm_i(m,      (int)dpk.u);   // {d_p0, d_p1} of element m
    e1.u = (unsigned)bperm_i(m + 16, (int)dpk.u);   // {d_p0, d_p1} of element m+16
    const h2f d0a = {e0.h[0], e0.h[0]}, d0b = {e0.h[1], e0.h[1]};
    const h2f d1a = {e1.h[0], e1.h[0]}, d1b = {e1.h[1], e1.h[1]};

    // layer 1 (packed f16): K 0..15 region <- pair p0, K 16..31 region <- pair p1
    union { unsigned u[8]; v16h h; } BE0, BE1;
    #pragma unroll
    for (int j = 0; j < 4; ++j) {
      H2U ta0, tb0, ta1, tb1;
      ta0.h = relu2(d0a * WaS[j].h + baS[j].h);   // v_pk_fma_f16 + v_pk_max
      tb0.h = relu2(d0b * WaS[j].h + baS[j].h);
      ta1.h = relu2(d1a * WaS[j].h + baS[j].h);
      tb1.h = relu2(d1b * WaS[j].h + baS[j].h);
      BE0.u[j] = ta0.u;  BE0.u[j + 4] = tb0.u;
      BE1.u[j] = ta1.u;  BE1.u[j + 4] = tb1.u;
    }

    // layer 2: one fully-utilized WMMA per 16 elements covers BOTH pairs
    v8f cE0 = __builtin_amdgcn_wmma_f32_16x16x32_f16(
        false, afragC, false, BE0.h, (short)0, cb, false, false);  // elems 0..15
    v8f cE1 = __builtin_amdgcn_wmma_f32_16x16x32_f16(
        false, afragC, false, BE1.h, (short)0, cb, false, false);  // elems 16..31

    // layer 3 dots. Low lanes' rows = p0; high lanes' rows = p1.
    float magE0 = b3cr, magE1 = b3cr;
    #pragma unroll
    for (int o = 0; o < 8; ++o) {
      magE0 = fmaf(W3cr[o], fmaxf(cE0[o], 0.0f), magE0);
      magE1 = fmaf(W3cr[o], fmaxf(cE1[o], 0.0f), magE1);
    }
    // route: low lane m has (p0,own)=magE0, needs (p1,own) from lane m+16's magE0;
    //        high lane m+16 has (p1,own)=magE1, needs (p0,own) from lane m's magE1.
    const float tP0 = bperm_f(m,      magE1);   // (p0, elem m+16) -> high lanes
    const float tP1 = bperm_f(m + 16, magE0);   // (p1, elem m)    -> low lanes
    const float magP0 = hiHalf ? tP0 : magE0;
    const float magP1 = hiHalf ? magE1 : tP1;

    // symmetric accumulation for both pairs
    const float s0 = magP0 * rinv0;
    f3x[i0] += s0 * dx0; f3y[i0] += s0 * dy0;
    f3x[k0] -= s0 * dx0; f3y[k0] -= s0 * dy0;
    const float s1 = magP1 * rinv1;
    f3x[i1] += s1 * dx1; f3y[i1] += s1 * dy1;
    f3x[k1] -= s1 * dx1; f3y[k1] -= s1 * dy1;
  }

  // ---- f1 (door attraction), f2 (wall repulsion), combine, store ----
  if (inb) {
    float acc[2 * NP];
    #pragma unroll
    for (int p = 0; p < NP; ++p) {
      const float dx = 5.0f - Xx[p], dy = 0.0f - Xy[p];
      const float rn = __frsqrt_rn(dx * dx + dy * dy);
      const float ux = dx * rn, uy = dy * rn;
      const float f1x = fmaf(W1r[0], ux, fmaf(W1r[1], uy, fmaf(W1r[2], Vx[p], fmaf(W1r[3], Vy[p], b1r[0]))));
      const float f1y = fmaf(W1r[4], ux, fmaf(W1r[5], uy, fmaf(W1r[6], Vx[p], fmaf(W1r[7], Vy[p], b1r[1]))));

      const float x0 = Xx[p], x1 = Xy[p];
      const float rd = (x1 < 1.0f && x1 > -1.0f) ? 100.0f : (5.0f - x0);
      const float fx = __expf(fmaf(w2r, x0 + 5.0f, b2r)) - __expf(fmaf(w2r, rd, b2r));
      const float fy = __expf(fmaf(w2r, x1 + 5.0f, b2r)) - __expf(fmaf(w2r, 5.0f - x1, b2r));

      acc[2 * p]     = (f1x + fx + f3x[p]) * 0.0125f;   // /80
      acc[2 * p + 1] = (f1y + fy + f3y[p]) * 0.0125f;
    }
    float4* ov = (float4*)(out + (size_t)tid * 20);
    ov[0] = make_float4(Vx[0], Vy[0], Vx[1], Vy[1]);
    ov[1] = make_float4(Vx[2], Vy[2], Vx[3], Vy[3]);
    ov[2] = make_float4(Vx[4], Vy[4], acc[0], acc[1]);
    ov[3] = make_float4(acc[2], acc[3], acc[4], acc[5]);
    ov[4] = make_float4(acc[6], acc[7], acc[8], acc[9]);
  }
}

extern "C" void kernel_launch(void* const* d_in, const int* in_sizes, int n_in,
                              void* d_out, int out_size, void* d_ws, size_t ws_size,
                              hipStream_t stream) {
  // setup_inputs order: t, y, W1, b1, w2, b2, W3a, b3a, W3b, b3b, W3c, b3c
  const float* y   = (const float*)d_in[1];
  const float* W1  = (const float*)d_in[2];
  const float* b1  = (const float*)d_in[3];
  const float* w2  = (const float*)d_in[4];
  const float* b2  = (const float*)d_in[5];
  const float* W3a = (const float*)d_in[6];
  const float* b3a = (const float*)d_in[7];
  const float* W3b = (const float*)d_in[8];
  const float* b3b = (const float*)d_in[9];
  const float* W3c = (const float*)d_in[10];
  const float* b3c = (const float*)d_in[11];
  float* out = (float*)d_out;

  const int nB = in_sizes[1] / (4 * NP);          // 262144
  const int threads = 256;                         // 8 wave32 per block
  const int blocks = (nB + threads - 1) / threads;
  odefunc_kernel<<<blocks, threads, 0, stream>>>(y, W1, b1, w2, b2, W3a, b3a,
                                                 W3b, b3b, W3c, b3c, out, nB);
}
